// TripartiteHeteroGNN_27212912788269
// MI455X (gfx1250) — matrix operand order repaired and checked by reading, not compiled, not run on hardware
//
#include <hip/hip_runtime.h>
#include <hip/hip_bf16.h>
#include <stdint.h>

// ---------------------------------------------------------------------------
// TripartiteHeteroGNN forward for MI455X (gfx1250), wave32 + WMMA.
//
// Roofline: ~45 G MACs of GEMM (negligible on the WMMA pipes at 16x16x32
// bf16/instr) vs ~2.6 GB of edge gather/scatter traffic (~0.1-0.2 ms at
// 23.3 TB/s) -> all dense math runs through v_wmma_f32_16x16x32_bf16 and the
// edge kernels are pure streaming loads + f32 global atomics.
// ---------------------------------------------------------------------------

typedef __attribute__((ext_vector_type(8)))  float    v8f;
typedef __attribute__((ext_vector_type(16))) __bf16   v16bf;
typedef __attribute__((ext_vector_type(8)))  unsigned v8u;

#define HID 128

// f32 -> bf16 (round-to-nearest, ties-away: u + 0x8000), scalar form.
__device__ __forceinline__ __bf16 f2bf(float f) {
  unsigned u = __builtin_bit_cast(unsigned, f) + 0x8000u;
  unsigned short h = (unsigned short)(u >> 16);
  return __builtin_bit_cast(__bf16, h);
}

// Pack two f32 into one dword of two bf16 using a single v_perm_b32:
// low half <- round(f0).bytes[3:2], high half <- round(f1).bytes[3:2].
__device__ __forceinline__ unsigned pack2_bf16(float f0, float f1) {
  unsigned u0 = __builtin_bit_cast(unsigned, f0) + 0x8000u;
  unsigned u1 = __builtin_bit_cast(unsigned, f1) + 0x8000u;
  return __builtin_amdgcn_perm(u1, u0, 0x07060302u);
}

// ---------------------------------------------------------------------------
// Weight prep: W [K x 128] f32 (row major, reference computes x @ W) ->
// WT [128 x K] bf16 so a B-fragment per lane is one contiguous 32B load.
// ---------------------------------------------------------------------------
__global__ void prep_wt_kernel(const float* __restrict__ W, __bf16* __restrict__ WT, int K) {
  int t = blockIdx.x * blockDim.x + threadIdx.x;
  if (t >= K * HID) return;
  int n = t / K;          // output row (N dim)
  int k = t - n * K;      // output col (K dim)
  WT[t] = f2bf(W[k * HID + n]);
}

__global__ void zero_f32_kernel(float* __restrict__ p, int n) {
  int t = blockIdx.x * blockDim.x + threadIdx.x;
  if (t < n) p[t] = 0.0f;
}

// ---------------------------------------------------------------------------
// WMMA GEMM:  C[M x 128] = epilogue( A[M x K] @ W[K x 128] + bias )
//   epilogue: v = (1+*epsPtr) * (acc + bias)  (+ Cadd[row,col])  (relu?)
//   C written at C[row*ldc + colOff + col]   (for 'cat' outputs, ldc=256)
//
// One wave computes a 16-row x 128-col tile; 8 waves/block -> 128 rows/block.
// A fragments (bf16, contiguous K order: lanes 16..31 at K+16, VGPRs 4..7 at
// K+8) are register-resident across all eight 16-col N-tiles. Per N-tile,
// ALL B fragments are preloaded as one load clause before the WMMA chain so
// the per-wave schedule is loads-in-flight -> staged s_wait_loadcnt -> wmma
// (B layout: lanes 16..31 at K+16, VGPR v half h -> K=2v+h, contiguous 32B
// from the pre-transposed bf16 weights).
// Full tiles take a wave-uniform unguarded fast path.
// ---------------------------------------------------------------------------
template <int K>
__global__ __launch_bounds__(256) void gemm_wmma_kernel(
    const float* __restrict__ A, const __bf16* __restrict__ WT,
    const float* __restrict__ bias, const float* __restrict__ epsPtr,
    const float* __restrict__ Cadd, float* __restrict__ C,
    int M, int ldc, int colOff, int relu) {
  constexpr int KS = K / 32;
  const int wave = threadIdx.x >> 5;
  const int lane = threadIdx.x & 31;
  const int lh = lane & 15;
  const int hi = lane >> 4;
  const int tileRow = blockIdx.x * 128 + wave * 16;
  if (tileRow >= M) return;                 // uniform per wave -> EXEC all-ones at WMMA
  const bool fullTile = (tileRow + 16 <= M);  // uniform per wave

  const int arow = tileRow + lh;
  const bool rowOK = fullTile | (arow < M);
  const float* ap = A + (size_t)arow * K;

  // ---- load all A fragments for this 16-row strip ----
  v16bf afrag[KS];
#pragma unroll
  for (int ks = 0; ks < KS; ++ks) {
    const int kb = ks * 32 + hi * 16;  // lanes 16..31 handle K+16 (contiguous, matches B)
    float4 c00, c01, c10, c11;
    if (rowOK) {
      c00 = *(const float4*)(ap + kb);
      c01 = *(const float4*)(ap + kb + 4);
      c10 = *(const float4*)(ap + kb + 8);  // VGPRs 4..7 hold K+8 (contiguous)
      c11 = *(const float4*)(ap + kb + 12);
    } else {
      c00 = make_float4(0.f, 0.f, 0.f, 0.f);
      c01 = c00; c10 = c00; c11 = c00;
    }
    v8u pk;
    pk[0] = pack2_bf16(c00.x, c00.y);
    pk[1] = pack2_bf16(c00.z, c00.w);
    pk[2] = pack2_bf16(c01.x, c01.y);
    pk[3] = pack2_bf16(c01.z, c01.w);
    pk[4] = pack2_bf16(c10.x, c10.y);
    pk[5] = pack2_bf16(c10.z, c10.w);
    pk[6] = pack2_bf16(c11.x, c11.y);
    pk[7] = pack2_bf16(c11.z, c11.w);
    afrag[ks] = __builtin_bit_cast(v16bf, pk);
  }

  const float alpha = epsPtr ? (1.0f + epsPtr[0]) : 1.0f;

#pragma unroll
  for (int nt = 0; nt < 8; ++nt) {
    const __bf16* wp = WT + (size_t)(nt * 16 + lh) * K + hi * 16;

    // Preload ALL B fragments for this N-tile: one load clause; the WMMA
    // chain below then drains them with staged (non-zero) s_wait_loadcnt.
    v16bf bfrag[KS];
#pragma unroll
    for (int ks = 0; ks < KS; ++ks)
      bfrag[ks] = *(const v16bf*)(wp + ks * 32);

    v8f acc = {};
#pragma unroll
    for (int ks = 0; ks < KS; ++ks)
      acc = __builtin_amdgcn_wmma_f32_16x16x32_bf16(
          false, afrag[ks], false, bfrag[ks], (short)0, acc, false, false);

    const int ncol = nt * 16 + lh;
    const float b = bias ? bias[ncol] : 0.0f;
    float* cp = C + (size_t)tileRow * ldc + colOff + ncol;
    const float* xp = Cadd ? (Cadd + (size_t)tileRow * HID + ncol) : nullptr;
    const int rbase = hi * 8;  // C/D layout: VGPR r -> row r / r+8, N = lh
    if (fullTile) {
#pragma unroll
      for (int r = 0; r < 8; ++r) {
        float v = alpha * (acc[r] + b);
        if (Cadd) v += xp[(size_t)(r + rbase) * HID];
        if (relu) v = fmaxf(v, 0.0f);
        cp[(size_t)(r + rbase) * ldc] = v;
      }
    } else {
#pragma unroll
      for (int r = 0; r < 8; ++r) {
        const int row = tileRow + r + rbase;
        if (row < M) {
          float v = alpha * (acc[r] + b);
          if (Cadd) v += xp[(size_t)(r + rbase) * HID];
          if (relu) v = fmaxf(v, 0.0f);
          cp[(size_t)(r + rbase) * ldc] = v;
        }
      }
    }
  }
}

// ---------------------------------------------------------------------------
// Small-K linear + relu (encoder first layers, K in {1,2,4,6}).
// One thread per output element; W reads are fully coalesced.
// ---------------------------------------------------------------------------
__global__ void lin_small_relu_kernel(const float* __restrict__ X, int N, int K,
                                      const float* __restrict__ W,
                                      const float* __restrict__ b,
                                      float* __restrict__ Y) {
  int t = blockIdx.x * blockDim.x + threadIdx.x;
  if (t >= N * HID) return;
  int row = t >> 7;
  int col = t & (HID - 1);
  float acc = b[col];
  for (int k = 0; k < K; ++k) acc += X[row * K + k] * W[k * HID + col];
  Y[t] = fmaxf(acc, 0.0f);
}

// ---------------------------------------------------------------------------
// Edge message + scatter-sum:
//   msg = relu(S[src[e]] + ea[e]*We + be);  atomicAdd(agg[dst[e]], msg)
// 32 lanes per edge, 4 features per lane (float4 streaming loads).
// ---------------------------------------------------------------------------
__global__ __launch_bounds__(256) void edge_scatter_kernel(
    const float* __restrict__ S,
    const int* __restrict__ src,
    const int* __restrict__ dst,
    const float* __restrict__ ea,
    const float* __restrict__ We,
    const float* __restrict__ be,
    float* __restrict__ agg, int E) {
  int gid = blockIdx.x * blockDim.x + threadIdx.x;
  int e = gid >> 5;
  if (e >= E) return;
  int lane = gid & 31;
  int c = lane * 4;
  float a = ea[e];
  int s = src[e];
  int d = dst[e];
  float4 sv = *(const float4*)(S + (size_t)s * HID + c);
  float4 w = *(const float4*)(We + c);
  float4 bb = *(const float4*)(be + c);
  float m0 = fmaxf(sv.x + a * w.x + bb.x, 0.0f);
  float m1 = fmaxf(sv.y + a * w.y + bb.y, 0.0f);
  float m2 = fmaxf(sv.z + a * w.z + bb.z, 0.0f);
  float m3 = fmaxf(sv.w + a * w.w + bb.w, 0.0f);
  float* ap = agg + (size_t)d * HID + c;
  atomicAdd(ap + 0, m0);
  atomicAdd(ap + 1, m1);
  atomicAdd(ap + 2, m2);
  atomicAdd(ap + 3, m3);
}

// ---------------------------------------------------------------------------
// pred.l2: out[row] = P[row,:] . W2[:,0] + b2   (128-dot per thread)
// ---------------------------------------------------------------------------
__global__ void pred_out_kernel(const float* __restrict__ P,
                                const float* __restrict__ W2,
                                const float* __restrict__ b2,
                                float* __restrict__ out, int N) {
  int row = blockIdx.x * blockDim.x + threadIdx.x;
  if (row >= N) return;
  const float* p = P + (size_t)row * HID;
  float acc = b2[0];
#pragma unroll 4
  for (int j = 0; j < HID; j += 4) {
    float4 x = *(const float4*)(p + j);
    float4 w = *(const float4*)(W2 + j);
    acc += x.x * w.x + x.y * w.y + x.z * w.z + x.w * w.w;
  }
  out[row] = acc;
}

// ---------------------------------------------------------------------------
// Host-side orchestration
// ---------------------------------------------------------------------------
extern "C" void kernel_launch(void* const* d_in, const int* in_sizes, int n_in,
                              void* d_out, int out_size, void* d_ws, size_t ws_size,
                              hipStream_t stream) {
  // ---- input indices (setup_inputs insertion order) ----
  enum {
    IN_X_CONS = 0, IN_X_VALS, IN_X_OBJ, IN_X_START,
    IN_C2V_SRC, IN_C2V_DST, IN_V2C_SRC, IN_V2C_DST,
    IN_V2O_SRC, IN_V2O_DST, IN_O2V_SRC, IN_O2V_DST,
    IN_C2O_SRC, IN_C2O_DST, IN_O2C_SRC, IN_O2C_DST,
    IN_EA_C2V, IN_EA_V2C, IN_EA_V2O, IN_EA_O2V, IN_EA_C2O, IN_EA_O2C
  };
  // params pytree flattened in JAX order (dict keys sorted, lists in order):
  //   enc_cons(22) enc_obj(26) enc_start(30) enc_vals(34)  [each: l1.W,l1.b,l2.W,l2.b]
  //   layers(38): layer0 then layer1; relations sorted: c2o,c2v,o2c,o2v,v2c,v2o
  //     per conv (sorted): dst.W,dst.b, edge.W,edge.b, eps, l1.W,l1.b,l2.W,l2.b, src.W,src.b
  //   pred(170): l1.W,l1.b,l2.W,l2.b
  const int P_ENC_CONS = 22, P_ENC_OBJ = 26, P_ENC_START = 30, P_ENC_VALS = 34;
  const int P_LAYERS = 38, CONV_STRIDE = 11, LAYER_STRIDE = 66, P_PRED = 170;
  const int R_C2O = 0, R_C2V = 1, R_O2C = 2, R_O2V = 3, R_V2C = 4, R_V2O = 5;

  auto F = [&](int i) { return (const float*)d_in[i]; };
  auto I = [&](int i) { return (const int*)d_in[i]; };

  const int NC = in_sizes[IN_X_CONS] / 4;
  const int NV = in_sizes[IN_X_VALS] / 6;
  const int NO = in_sizes[IN_X_OBJ] / 2;
  const int E_CV = in_sizes[IN_C2V_SRC];
  const int E_VO = in_sizes[IN_V2O_SRC];
  const int E_CO = in_sizes[IN_C2O_SRC];

  // ---- workspace carving (256B aligned) ----
  size_t off = 0;
  auto carve = [&](size_t bytes) -> void* {
    void* p = (char*)d_ws + off;
    off += (bytes + 255) & ~(size_t)255;
    return p;
  };
  float* xc   = (float*)carve((size_t)NC * 128 * 4);
  float* xv   = (float*)carve((size_t)NV * 256 * 4);   // encoder 'cat'; reused as hv in layer 1
  float* xo   = (float*)carve((size_t)NO * 128 * 4);
  float* rnv  = (float*)carve((size_t)NV * 256 * 4);   // relu(layer0 vals)
  float* rnc  = (float*)carve((size_t)NC * 256 * 4);
  float* rno  = (float*)carve((size_t)NO * 256 * 4);
  float* bufS = (float*)carve((size_t)NV * 128 * 4);   // lin_src output
  float* agg  = (float*)carve((size_t)NV * 128 * 4);   // scatter accumulator
  float* bufH = (float*)carve((size_t)NV * 128 * 4);   // (1+eps)*lin_dst + agg
  float* bufM = (float*)carve((size_t)NV * 128 * 4);   // mlp hidden / scratch
  __bf16* wt  = (__bf16*)carve((size_t)256 * 128 * 2); // transposed bf16 weights
  float* hv   = xv;  // layer-1 'cat' output overlays xv (xv dead after layer 0)
  (void)ws_size; (void)n_in; (void)out_size;

  // ---- launch helpers ----
  auto gemm = [&](int K, const float* A, const float* W, const float* bias,
                  const float* epsPtr, const float* Cadd, float* C,
                  int M, int ldc, int colOff, int relu) {
    int tot = K * 128;
    prep_wt_kernel<<<(tot + 255) / 256, 256, 0, stream>>>(W, wt, K);
    int grid = (M + 127) / 128;
    if (K == 128)
      gemm_wmma_kernel<128><<<grid, 256, 0, stream>>>(A, wt, bias, epsPtr, Cadd, C, M, ldc, colOff, relu);
    else
      gemm_wmma_kernel<256><<<grid, 256, 0, stream>>>(A, wt, bias, epsPtr, Cadd, C, M, ldc, colOff, relu);
  };

  auto gine = [&](int pb, const float* xsrc, int Ns, int Ks,
                  const float* xdst, int Nd, int Kd,
                  const int* src, const int* dst, const float* ea, int E,
                  float* outbuf, int colOff, int relu) {
    // s = lin_src(x_src)
    gemm(Ks, xsrc, F(pb + 9), F(pb + 10), nullptr, nullptr, bufS, Ns, 128, 0, 0);
    // agg = segment_sum(relu(s[src] + ea*We + be), dst)
    int aggN = Nd * 128;
    zero_f32_kernel<<<(aggN + 255) / 256, 256, 0, stream>>>(agg, aggN);
    long ethreads = (long)E * 32;
    edge_scatter_kernel<<<(int)((ethreads + 255) / 256), 256, 0, stream>>>(
        bufS, src, dst, ea, F(pb + 2), F(pb + 3), agg, E);
    // h = (1+eps)*lin_dst(x_dst) + agg    (fused GEMM epilogue)
    gemm(Kd, xdst, F(pb + 0), F(pb + 1), F(pb + 4), agg, bufH, Nd, 128, 0, 0);
    // out = l2(relu(l1(h)))
    gemm(128, bufH, F(pb + 5), F(pb + 6), nullptr, nullptr, bufM, Nd, 128, 0, 1);
    gemm(128, bufM, F(pb + 7), F(pb + 8), nullptr, nullptr, outbuf, Nd, 256, colOff, relu);
  };

  auto enc = [&](int pb, const float* X, int N, int Kin, float* outp, int ldc, int colOff) {
    int tot = N * 128;
    lin_small_relu_kernel<<<(tot + 255) / 256, 256, 0, stream>>>(X, N, Kin, F(pb + 0), F(pb + 1), bufM);
    gemm(128, bufM, F(pb + 2), F(pb + 3), nullptr, nullptr, outp, N, ldc, colOff, 0);
  };

  // ---- encoders ----
  enc(P_ENC_CONS,  F(IN_X_CONS),  NC, 4, xc, 128, 0);
  enc(P_ENC_VALS,  F(IN_X_VALS),  NV, 6, xv, 256, 0);
  enc(P_ENC_START, F(IN_X_START), NV, 1, xv, 256, 128);
  enc(P_ENC_OBJ,   F(IN_X_OBJ),   NO, 2, xo, 128, 0);

  // ---- layer 0 (all six relations; outputs relu'd into rnv/rnc/rno) ----
  const int L0 = P_LAYERS;
  gine(L0 + R_C2V * CONV_STRIDE, xc, NC, 128, xv, NV, 256,
       I(IN_C2V_SRC), I(IN_C2V_DST), F(IN_EA_C2V), E_CV, rnv, 0, 1);
  gine(L0 + R_O2V * CONV_STRIDE, xo, NO, 128, xv, NV, 256,
       I(IN_O2V_SRC), I(IN_O2V_DST), F(IN_EA_O2V), E_VO, rnv, 128, 1);
  gine(L0 + R_V2C * CONV_STRIDE, xv, NV, 256, xc, NC, 128,
       I(IN_V2C_SRC), I(IN_V2C_DST), F(IN_EA_V2C), E_CV, rnc, 0, 1);
  gine(L0 + R_O2C * CONV_STRIDE, xo, NO, 128, xc, NC, 128,
       I(IN_O2C_SRC), I(IN_O2C_DST), F(IN_EA_O2C), E_CO, rnc, 128, 1);
  gine(L0 + R_V2O * CONV_STRIDE, xv, NV, 256, xo, NO, 128,
       I(IN_V2O_SRC), I(IN_V2O_DST), F(IN_EA_V2O), E_VO, rno, 0, 1);
  gine(L0 + R_C2O * CONV_STRIDE, xc, NC, 128, xo, NO, 128,
       I(IN_C2O_SRC), I(IN_C2O_DST), F(IN_EA_C2O), E_CO, rno, 128, 1);

  // ---- layer 1: only vals-targeting relations reach the output ----
  const int L1 = P_LAYERS + LAYER_STRIDE;
  gine(L1 + R_C2V * CONV_STRIDE, rnc, NC, 256, rnv, NV, 256,
       I(IN_C2V_SRC), I(IN_C2V_DST), F(IN_EA_C2V), E_CV, hv, 0, 0);
  gine(L1 + R_O2V * CONV_STRIDE, rno, NO, 256, rnv, NV, 256,
       I(IN_O2V_SRC), I(IN_O2V_DST), F(IN_EA_O2V), E_VO, hv, 128, 0);

  // ---- prediction head ----
  gemm(256, hv, F(P_PRED + 0), F(P_PRED + 1), nullptr, nullptr, bufM, NV, 128, 0, 1);
  pred_out_kernel<<<(NV + 255) / 256, 256, 0, stream>>>(
      bufM, F(P_PRED + 2), F(P_PRED + 3), (float*)d_out, NV);
}
